// GRANet_69432441307815
// MI455X (gfx1250) — matrix operand order repaired and checked by reading, not compiled, not run on hardware
//
#include <hip/hip_runtime.h>
#include <hip/hip_bf16.h>

typedef _Float16 h16v __attribute__((ext_vector_type(16)));
typedef _Float16 h8v  __attribute__((ext_vector_type(8)));
typedef float    f8v  __attribute__((ext_vector_type(8)));

#define IN_F   128
#define HID_F  64
#define HEADS  4
#define HO_F   256
#define OUT_F  2
#define NGRAPH 256
#define TB     256

static inline int cdiv_i(long a, long b){ return (int)((a + b - 1) / b); }

// ---------------- utility ----------------
__global__ void k_fill_f32(float* __restrict__ p, float v, long n){
  long i = (long)blockIdx.x * blockDim.x + threadIdx.x;
  if (i < n) p[i] = v;
}
__global__ void k_fill_u32(unsigned* __restrict__ p, unsigned v, long n){
  long i = (long)blockIdx.x * blockDim.x + threadIdx.x;
  if (i < n) p[i] = v;
}
__global__ void k_cvt_f16(const float* __restrict__ s, _Float16* __restrict__ d, long n){
  long i = (long)blockIdx.x * blockDim.x + threadIdx.x;
  if (i < n) d[i] = (_Float16)s[i];
}

__device__ __forceinline__ unsigned fenc(float f){
  unsigned u = __float_as_uint(f);
  return (u & 0x80000000u) ? ~u : (u | 0x80000000u);
}
__device__ __forceinline__ float fdec(unsigned u){
  return (u & 0x80000000u) ? __uint_as_float(u & 0x7FFFFFFFu) : __uint_as_float(~u);
}

// ---------------- degrees ----------------
__global__ void k_deg(const int* __restrict__ dst, unsigned* __restrict__ deg, int E){
  int i = blockIdx.x * blockDim.x + threadIdx.x;
  if (i < E) atomicAdd(&deg[dst[i]], 1u);
}
__global__ void k_dinv(const unsigned* __restrict__ deg, float* __restrict__ dinv, int n){
  int i = blockIdx.x * blockDim.x + threadIdx.x;
  if (i < n){ unsigned u = deg[i]; if (u < 1u) u = 1u; dinv[i] = rsqrtf((float)u); }
}

// ---------------- B-fragment packer ----------------
// Converts f32 weight B[K,Nc] into WMMA b-fragment layout:
//   P[((kt*NT + nt)*32 + lane)*16 + i] = (f16) B[(kt*32 + 8*(lane>>4) + (i<8 ? i : i+8))*Nc + nt*16 + (lane&15)]
// so each lane's 16-half fragment is 32 contiguous bytes.
__global__ void k_pack_b(const float* __restrict__ B, _Float16* __restrict__ P, int K, int Nc){
  int gid = blockIdx.x * blockDim.x + threadIdx.x;
  if (gid >= K * Nc) return;
  int i    = gid & 15;
  int lane = (gid >> 4) & 31;
  int t    = gid >> 9;
  int NT = Nc >> 4;
  int kt = t / NT;
  int nt = t - kt * NT;
  int kk = ((lane & 16) ? 8 : 0) + (i < 8 ? i : i + 8);
  int row = kt * 32 + kk;
  int col = nt * 16 + (lane & 15);
  P[gid] = (_Float16)B[(size_t)row * Nc + col];
}

// ---------------- WMMA GEMM: C[M,Nc] = A[M,K](f16) * Bp(packed f16) (+bias) ----------------
// Compile-time K and Nc => all fragment addresses are immediate offsets from two base
// pointers, fully unrolled k-loop, 1 A-frag feeds NTILES WMMAs.
template<int K, int Nc, int NTILES>
__global__ __launch_bounds__(TB)
void k_wmma_gemm(const _Float16* __restrict__ A, const _Float16* __restrict__ Bp,
                 float* __restrict__ C, const float* __restrict__ bias, int M)
{
  constexpr int NT      = Nc >> 4;
  constexpr int nGroups = NT / NTILES;
  const int lane  = threadIdx.x & 31;
  const int wave  = threadIdx.x >> 5;
  const int numMT = (M + 15) >> 4;
  const int tile  = blockIdx.x * (blockDim.x >> 5) + wave;
  if (tile >= numMT * nGroups) return;        // wave-uniform: EXEC all-1 for WMMA
  const int mt = tile / nGroups;
  const int ng = tile - mt * nGroups;

  int mA = (mt << 4) + (lane & 15);
  if (mA >= M) mA = M - 1;
  const int kA0 = (lane & 16) ? 8 : 0;

  const _Float16* abase = A + (size_t)mA * K + kA0;
  const _Float16* bbase = Bp + ((size_t)(ng * NTILES) * 32 + lane) * 16;

  f8v c[NTILES];
#pragma unroll
  for (int t = 0; t < NTILES; ++t) c[t] = (f8v){0.f,0.f,0.f,0.f,0.f,0.f,0.f,0.f};

#pragma unroll
  for (int kt = 0; kt < K / 32; ++kt){
    h8v alo = *(const h8v*)(abase + kt * 32);
    h8v ahi = *(const h8v*)(abase + kt * 32 + 16);
    h16v a;
#pragma unroll
    for (int i = 0; i < 8; ++i){ a[i] = alo[i]; a[i + 8] = ahi[i]; }
#pragma unroll
    for (int t = 0; t < NTILES; ++t){
      const h16v b = *(const h16v*)(bbase + (size_t)(kt * NT + t) * 512);
      c[t] = __builtin_amdgcn_wmma_f32_16x16x32_f16(false, a, false, b, (short)0, c[t], false, false);
    }
  }

  const int mst = (mt << 4) + ((lane & 16) ? 8 : 0);
#pragma unroll
  for (int t = 0; t < NTILES; ++t){
    const int nB = (ng * NTILES + t) * 16 + (lane & 15);
    const float bv = bias ? bias[nB] : 0.f;
    float* cp = C + (size_t)mst * Nc + nB;
#pragma unroll
    for (int j = 0; j < 8; ++j)
      if (mst + j < M) cp[(size_t)j * Nc] = c[t][j] + bv;
  }
}

// ---------------- GCN normalized scatter-add (64-wide, float4 per thread) ----------------
__global__ void k_gcn_scatter(const int* __restrict__ src, const int* __restrict__ dst,
                              const float* __restrict__ dinv, const float* __restrict__ t,
                              float* __restrict__ acc, int E, int total /*EE*16*/)
{
  int gid = blockIdx.x * blockDim.x + threadIdx.x;
  if (gid >= total) return;
  int e = gid >> 4, f4 = gid & 15;
  int s, d;
  if (e < E){ s = src[e]; d = dst[e]; } else { s = d = e - E; }
  float nrm = dinv[s] * dinv[d];
  const float4 v = *(const float4*)(t + (size_t)s * HID_F + f4 * 4);
  float* ap = acc + (size_t)d * HID_F + f4 * 4;
  atomicAdd(ap + 0, v.x * nrm);
  atomicAdd(ap + 1, v.y * nrm);
  atomicAdd(ap + 2, v.z * nrm);
  atomicAdd(ap + 3, v.w * nrm);
}

__global__ void k_bias_relu(float* __restrict__ acc, const float* __restrict__ b, long n){
  long i = (long)blockIdx.x * blockDim.x + threadIdx.x;
  if (i < n){ float v = acc[i] + b[i & (HID_F - 1)]; acc[i] = v > 0.f ? v : 0.f; }
}
// h = relu(acc + b2) + residual  (residual held in hbuf, overwritten with h)
__global__ void k_combine_h(float* __restrict__ hbuf, const float* __restrict__ acc,
                            const float* __restrict__ b2, long n){
  long i = (long)blockIdx.x * blockDim.x + threadIdx.x;
  if (i < n){ float v = acc[i] + b2[i & (HID_F - 1)]; v = v > 0.f ? v : 0.f; hbuf[i] = hbuf[i] + v; }
}

// ---------------- attention ----------------
__global__ void k_asd(const float* __restrict__ hh, const float* __restrict__ att_src,
                      const float* __restrict__ att_dst, float* __restrict__ a_s,
                      float* __restrict__ a_d, int n /*N*HEADS*/)
{
  int gid = blockIdx.x * blockDim.x + threadIdx.x;
  if (gid >= n) return;
  int node = gid >> 2, h = gid & 3;
  const float* hp = hh + (size_t)node * HO_F + h * HID_F;
  const float* as = att_src + h * HID_F;
  const float* ad = att_dst + h * HID_F;
  float s0 = 0.f, s1 = 0.f;
  for (int c = 0; c < HID_F; ++c){ float v = hp[c]; s0 += v * as[c]; s1 += v * ad[c]; }
  a_s[gid] = s0; a_d[gid] = s1;
}

__global__ void k_edge_logit(const int* __restrict__ src, const int* __restrict__ dst,
                             const float* __restrict__ a_s, const float* __restrict__ a_d,
                             float* __restrict__ ebuf, unsigned* __restrict__ menc,
                             int E, int total /*EE*HEADS*/)
{
  int gid = blockIdx.x * blockDim.x + threadIdx.x;
  if (gid >= total) return;
  int e = gid >> 2, h = gid & 3;
  int s, d;
  if (e < E){ s = src[e]; d = dst[e]; } else { s = d = e - E; }
  float ev = a_s[(size_t)s * HEADS + h] + a_d[(size_t)d * HEADS + h];
  ev = ev > 0.f ? ev : 0.2f * ev;           // leaky_relu(.,0.2)
  ebuf[gid] = ev;
  atomicMax(&menc[(size_t)d * HEADS + h], fenc(ev));
}

__global__ void k_edge_exp(const int* __restrict__ dst, const unsigned* __restrict__ menc,
                           float* __restrict__ ebuf, float* __restrict__ denom,
                           int E, int total)
{
  int gid = blockIdx.x * blockDim.x + threadIdx.x;
  if (gid >= total) return;
  int e = gid >> 2, h = gid & 3;
  int d = (e < E) ? dst[e] : (e - E);
  float m = fdec(menc[(size_t)d * HEADS + h]);
  float ex = expf(ebuf[gid] - m);
  ebuf[gid] = ex;
  atomicAdd(&denom[(size_t)d * HEADS + h], ex);
}

__global__ void k_edge_alpha(const int* __restrict__ dst, const float* __restrict__ denom,
                             float* __restrict__ ebuf, int E, int total)
{
  int gid = blockIdx.x * blockDim.x + threadIdx.x;
  if (gid >= total) return;
  int e = gid >> 2, h = gid & 3;
  int d = (e < E) ? dst[e] : (e - E);
  ebuf[gid] = ebuf[gid] / (denom[(size_t)d * HEADS + h] + 1e-16f);
}

// ---------------- GAT weighted scatter (256-wide, float4 per thread) ----------------
__global__ void k_gat_scatter(const int* __restrict__ src, const int* __restrict__ dst,
                              const float* __restrict__ alpha, const float* __restrict__ hh,
                              float* __restrict__ acc, int E, int total /*EE*64*/)
{
  int gid = blockIdx.x * blockDim.x + threadIdx.x;
  if (gid >= total) return;
  int e = gid >> 6, f4 = gid & 63;
  int s, d;
  if (e < E){ s = src[e]; d = dst[e]; } else { s = d = e - E; }
  float al = alpha[(size_t)e * HEADS + (f4 >> 4)];
  const float4 v = *(const float4*)(hh + (size_t)s * HO_F + f4 * 4);
  float* ap = acc + (size_t)d * HO_F + f4 * 4;
  atomicAdd(ap + 0, v.x * al);
  atomicAdd(ap + 1, v.y * al);
  atomicAdd(ap + 2, v.z * al);
  atomicAdd(ap + 3, v.w * al);
}

// ---------------- pooling / head ----------------
__global__ void k_cnt(const int* __restrict__ batch, float* __restrict__ cnt, int n){
  int i = blockIdx.x * blockDim.x + threadIdx.x;
  if (i < n) atomicAdd(&cnt[batch[i]], 1.f);
}
__global__ void k_pool(const float* __restrict__ acc, const float* __restrict__ gat_b,
                       const int* __restrict__ batch, float* __restrict__ pooled, long n /*N*HO*/)
{
  long gid = (long)blockIdx.x * blockDim.x + threadIdx.x;
  if (gid >= n) return;
  int node = (int)(gid >> 8), f = (int)(gid & 255);
  float v = acc[gid] + gat_b[f];
  v = v > 0.f ? v : 0.f;
  atomicAdd(&pooled[(size_t)batch[node] * HO_F + f], v);
}
__global__ void k_head(const float* __restrict__ pooled, const float* __restrict__ cnt,
                       const float* __restrict__ fc_w, const float* __restrict__ fc_b,
                       float* __restrict__ out, int G)
{
  int g = blockIdx.x * blockDim.x + threadIdx.x;
  if (g >= G) return;
  float c = cnt[g]; if (c < 1.f) c = 1.f;
  float l0 = fc_b[0], l1 = fc_b[1];
  const float* pp = pooled + (size_t)g * HO_F;
  for (int f = 0; f < HO_F; ++f){
    float p = pp[f] / c;
    l0 += p * fc_w[f * OUT_F + 0];
    l1 += p * fc_w[f * OUT_F + 1];
  }
  float mx = fmaxf(l0, l1);
  float lse = mx + logf(expf(l0 - mx) + expf(l1 - mx));
  out[g * OUT_F + 0] = l0 - lse;
  out[g * OUT_F + 1] = l1 - lse;
}

// ---------------- host ----------------
extern "C" void kernel_launch(void* const* d_in, const int* in_sizes, int n_in,
                              void* d_out, int out_size, void* d_ws, size_t ws_size,
                              hipStream_t stream)
{
  (void)n_in; (void)out_size; (void)ws_size;
  const float* x       = (const float*)d_in[0];
  const int*   eidx    = (const int*)  d_in[1];
  const int*   batch   = (const int*)  d_in[2];
  const float* w1      = (const float*)d_in[3];
  const float* b1      = (const float*)d_in[4];
  const float* w2      = (const float*)d_in[5];
  const float* b2      = (const float*)d_in[6];
  const float* gat_w   = (const float*)d_in[7];
  const float* att_src = (const float*)d_in[8];
  const float* att_dst = (const float*)d_in[9];
  const float* gat_b   = (const float*)d_in[10];
  const float* res_w   = (const float*)d_in[11];
  const float* res_b   = (const float*)d_in[12];
  const float* fc_w    = (const float*)d_in[13];
  const float* fc_b    = (const float*)d_in[14];

  const int N  = in_sizes[0] / IN_F;
  const int E  = in_sizes[1] / 2;
  const int EE = E + N;
  const int G  = NGRAPH;

  const int* srcI = eidx;
  const int* dstI = eidx + E;

  float* logits = (float*)d_out;
  float* alpha  = (float*)d_out + (size_t)G * OUT_F;   // [EE, HEADS], also used as e/ex scratch

  // ---- carve workspace ----
  char* base = (char*)d_ws;
  size_t off = 0;
  auto carve = [&](size_t bytes) -> void* {
    void* p = base + off; off += (bytes + 255) & ~(size_t)255; return p;
  };
  unsigned* deg    = (unsigned*)carve((size_t)N * 4);          // becomes dinv (f32) in place
  float*    dinv   = (float*)deg;
  unsigned* menc   = (unsigned*)carve((size_t)N * HEADS * 4);
  float*    denom  = (float*)   carve((size_t)N * HEADS * 4);
  float*    cnt    = (float*)   carve((size_t)G * 4);
  float*    pooled = (float*)   carve((size_t)G * HO_F * 4);
  float*    a_s    = (float*)   carve((size_t)N * HEADS * 4);
  float*    a_d    = (float*)   carve((size_t)N * HEADS * 4);
  _Float16* w1p    = (_Float16*)carve((size_t)IN_F  * HID_F * 2);  // packed b-fragments
  _Float16* w2p    = (_Float16*)carve((size_t)HID_F * HID_F * 2);
  _Float16* gwp    = (_Float16*)carve((size_t)HID_F * HO_F  * 2);
  _Float16* rwp    = (_Float16*)carve((size_t)IN_F  * HID_F * 2);
  _Float16* ah     = (_Float16*)carve((size_t)N * IN_F * 2);   // x_f16, reused for h1_f16 / h_f16
  float*    buf1   = (float*)   carve((size_t)N * HID_F * 4);  // residual -> h
  float*    buf2   = (float*)   carve((size_t)N * HID_F * 4);  // t1 -> t2
  float*    buf3   = (float*)   carve((size_t)N * HID_F * 4);  // h1acc -> h1 -> h2acc
  float*    hh     = (float*)   carve((size_t)N * HO_F * 4);
  float*    outacc = (float*)   carve((size_t)N * HO_F * 4);

  // ---- degrees (self-loop => init 1) ----
  k_fill_u32<<<cdiv_i(N, TB), TB, 0, stream>>>(deg, 1u, N);
  k_deg<<<cdiv_i(E, TB), TB, 0, stream>>>(dstI, deg, E);
  k_dinv<<<cdiv_i(N, TB), TB, 0, stream>>>(deg, dinv, N);

  // ---- f16 conversions / weight packing ----
  k_cvt_f16<<<cdiv_i((long)N * IN_F, TB), TB, 0, stream>>>(x, ah, (long)N * IN_F);
  k_pack_b<<<cdiv_i(IN_F * HID_F, TB), TB, 0, stream>>>(w1, w1p, IN_F, HID_F);
  k_pack_b<<<cdiv_i(HID_F * HID_F, TB), TB, 0, stream>>>(w2, w2p, HID_F, HID_F);
  k_pack_b<<<cdiv_i(HID_F * HO_F, TB), TB, 0, stream>>>(gat_w, gwp, HID_F, HO_F);
  k_pack_b<<<cdiv_i(IN_F * HID_F, TB), TB, 0, stream>>>(res_w, rwp, IN_F, HID_F);

  const int wavesPerBlk = TB / 32;
  // NTILES = 4 => each wave covers 64 output columns
  auto gemm_blocks = [&](int M, int Nc){
    return cdiv_i((long)((M + 15) / 16) * (Nc / 64), wavesPerBlk);
  };

  // ---- residual = x @ res_w + res_b ----
  k_wmma_gemm<IN_F, HID_F, 4><<<gemm_blocks(N, HID_F), TB, 0, stream>>>(ah, rwp, buf1, res_b, N);
  // ---- t1 = x @ w1 ----
  k_wmma_gemm<IN_F, HID_F, 4><<<gemm_blocks(N, HID_F), TB, 0, stream>>>(ah, w1p, buf2, nullptr, N);

  // ---- GCN layer 1 ----
  k_fill_f32<<<cdiv_i((long)N * HID_F, TB), TB, 0, stream>>>(buf3, 0.f, (long)N * HID_F);
  k_gcn_scatter<<<cdiv_i((long)EE * 16, TB), TB, 0, stream>>>(srcI, dstI, dinv, buf2, buf3, E, EE * 16);
  k_bias_relu<<<cdiv_i((long)N * HID_F, TB), TB, 0, stream>>>(buf3, b1, (long)N * HID_F);

  // ---- GCN layer 2 ----
  k_cvt_f16<<<cdiv_i((long)N * HID_F, TB), TB, 0, stream>>>(buf3, ah, (long)N * HID_F);
  k_wmma_gemm<HID_F, HID_F, 4><<<gemm_blocks(N, HID_F), TB, 0, stream>>>(ah, w2p, buf2, nullptr, N);
  k_fill_f32<<<cdiv_i((long)N * HID_F, TB), TB, 0, stream>>>(buf3, 0.f, (long)N * HID_F);
  k_gcn_scatter<<<cdiv_i((long)EE * 16, TB), TB, 0, stream>>>(srcI, dstI, dinv, buf2, buf3, E, EE * 16);
  k_combine_h<<<cdiv_i((long)N * HID_F, TB), TB, 0, stream>>>(buf1, buf3, b2, (long)N * HID_F);

  // ---- GAT projection hh = h @ gat_w ----
  k_cvt_f16<<<cdiv_i((long)N * HID_F, TB), TB, 0, stream>>>(buf1, ah, (long)N * HID_F);
  k_wmma_gemm<HID_F, HO_F, 4><<<gemm_blocks(N, HO_F), TB, 0, stream>>>(ah, gwp, hh, nullptr, N);

  // ---- attention coefficients ----
  k_asd<<<cdiv_i((long)N * HEADS, TB), TB, 0, stream>>>(hh, att_src, att_dst, a_s, a_d, N * HEADS);
  k_fill_u32<<<cdiv_i((long)N * HEADS, TB), TB, 0, stream>>>(menc, 0u, (long)N * HEADS);
  k_fill_f32<<<cdiv_i((long)N * HEADS, TB), TB, 0, stream>>>(denom, 0.f, (long)N * HEADS);
  k_edge_logit<<<cdiv_i((long)EE * HEADS, TB), TB, 0, stream>>>(srcI, dstI, a_s, a_d, alpha, menc, E, EE * HEADS);
  k_edge_exp<<<cdiv_i((long)EE * HEADS, TB), TB, 0, stream>>>(dstI, menc, alpha, denom, E, EE * HEADS);
  k_edge_alpha<<<cdiv_i((long)EE * HEADS, TB), TB, 0, stream>>>(dstI, denom, alpha, E, EE * HEADS);

  // ---- GAT aggregation ----
  k_fill_f32<<<cdiv_i((long)N * HO_F, TB), TB, 0, stream>>>(outacc, 0.f, (long)N * HO_F);
  k_gat_scatter<<<cdiv_i((long)EE * 64, TB), TB, 0, stream>>>(srcI, dstI, alpha, hh, outacc, E, EE * 64);

  // ---- global mean pool + FC + log_softmax ----
  k_fill_f32<<<cdiv_i((long)G * HO_F, TB), TB, 0, stream>>>(pooled, 0.f, (long)G * HO_F);
  k_fill_f32<<<cdiv_i((long)G, TB), TB, 0, stream>>>(cnt, 0.f, (long)G);
  k_cnt<<<cdiv_i(N, TB), TB, 0, stream>>>(batch, cnt, N);
  k_pool<<<cdiv_i((long)N * HO_F, TB), TB, 0, stream>>>(outacc, gat_b, batch, pooled, (long)N * HO_F);
  k_head<<<cdiv_i(G, TB), TB, 0, stream>>>(pooled, cnt, fc_w, fc_b, logits, G);
}
